// AirportGNN_69801808495357
// MI455X (gfx1250) — matrix operand order repaired and checked by reading, not compiled, not run on hardware
//
#include <hip/hip_runtime.h>
#include <hip/hip_bf16.h>

#define N_NODES 50000
#define N_EDGES 800000
#define HEADS   4
#define CH      24
#define HID     96
#define NZ      384   // BATCH * NUM_ZONES

typedef __attribute__((ext_vector_type(16))) _Float16 v16h;
typedef __attribute__((ext_vector_type(8)))  _Float16 v8h;
typedef __attribute__((ext_vector_type(8)))  float    v8f;

static inline int cdiv(int a, int b) { return (a + b - 1) / b; }

// order-preserving float <-> uint mapping for atomicMax on floats
__device__ __forceinline__ unsigned fenc(float f) {
    unsigned u = __float_as_uint(f);
    return (u & 0x80000000u) ? ~u : (u | 0x80000000u);
}
__device__ __forceinline__ float fdec(unsigned u) {
    return (u & 0x80000000u) ? __uint_as_float(u & 0x7FFFFFFFu) : __uint_as_float(~u);
}

__global__ void k_zero_f32(float* p, int n) {
    int i = blockIdx.x * blockDim.x + threadIdx.x;
    if (i < n) p[i] = 0.f;
}
__global__ void k_zero_u32(unsigned* p, int n) {
    int i = blockIdx.x * blockDim.x + threadIdx.x;
    if (i < n) p[i] = 0u;
}

// ---------------------------------------------------------------------------
// WMMA GEMM: C[M,Nn] = act(A[M,K] @ B[K,Nn] + bias)
// Compile-time K (mult of 32), NN (mult of 16); M must be a multiple of 16.
// Block = NN/16 waves (one 16-col tile each); MT row-tiles per block reusing
// the LDS-resident transposed B. Fragments are built with b128 LDS loads:
//   A row-major tile  -> two contiguous v8h runs per k-step
//   B transposed tile -> one contiguous v16h run per k-step
// ---------------------------------------------------------------------------
template<int K, int NN, int ACT, int MT>
__global__ __launch_bounds__((NN / 16) * 32)
void k_gemm_wmma(const float* __restrict__ A, const float* __restrict__ B,
                 const float* __restrict__ bias, float* __restrict__ C, int M)
{
    constexpr int NT = NN / 16;
    __shared__ __align__(32) _Float16 BT[NN * K];   // BT[n*K + k] = B[k*NN + n]
    __shared__ __align__(32) _Float16 Ah[16 * K];   // row-major 16 x K tile
    const int lane = threadIdx.x & 31;
    const int wave = threadIdx.x >> 5;              // n-tile index

    for (int i = threadIdx.x; i < K * NN; i += 32 * NT) {
        int n = i / K, k = i - n * K;
        BT[i] = (_Float16)B[k * NN + n];
    }

    const int n0   = wave * 16;
    const int m    = lane & 15;          // A row within tile / D col within tile
    const int aK   = (lane >> 4) * 16;   // A K-group split (contiguous, matches B)
    const int bK   = (lane >> 4) * 16;   // B K-group split
    const int ncol = n0 + m;
    const float bb = bias ? bias[ncol] : 0.f;

    for (int t = 0; t < MT; ++t) {
        const int row0 = (blockIdx.x * MT + t) * 16;
        if (row0 >= M) break;
        __syncthreads();                 // BT ready (t==0) / Ah consumed (t>0)
        for (int i = threadIdx.x; i < 16 * K; i += 32 * NT)
            Ah[i] = (_Float16)A[(long)row0 * K + i];   // 16 rows are contiguous
        __syncthreads();

        v8f acc = {};
#pragma unroll
        for (int k0 = 0; k0 < K; k0 += 32) {
            v8h a0 = *(const v8h*)&Ah[m * K + k0 + aK];
            v8h a1 = *(const v8h*)&Ah[m * K + k0 + aK + 8];
            v16h a;
#pragma unroll
            for (int j = 0; j < 8; ++j) { a[j] = a0[j]; a[j + 8] = a1[j]; }
            v16h b = *(const v16h*)&BT[ncol * K + k0 + bK];
            acc = __builtin_amdgcn_wmma_f32_16x16x32_f16(
                      false, a, false, b, (short)0, acc, false, false);
        }

        const int mbase = row0 + ((lane >> 4) * 8);  // D: row = r + 8*(lane>>4)
#pragma unroll
        for (int r = 0; r < 8; ++r) {
            float v = acc[r] + bb;
            if (ACT == 1) v = fmaxf(v, 0.f);
            C[(long)(mbase + r) * NN + ncol] = v;
        }
    }
}

// input MLP layer 1: h1 = relu(x[N,12] @ w1[12,96] + b1)   (K=12, memory bound)
__global__ void k_in_mlp1(const float* __restrict__ x, const float* __restrict__ w1,
                          const float* __restrict__ b1, float* __restrict__ h1, int n)
{
    int i = blockIdx.x * blockDim.x + threadIdx.x;
    if (i >= n * HID) return;
    int node = i / HID, f = i - node * HID;
    float s = b1[f];
#pragma unroll
    for (int k = 0; k < 12; ++k) s += x[node * 12 + k] * w1[k * HID + f];
    h1[i] = fmaxf(s, 0.f);
}

// degree + loop-attr segment sums (layer invariant)
__global__ void k_deg(const int* __restrict__ dst, const float* __restrict__ eattr,
                      float* __restrict__ deg, float* __restrict__ lsum, int E_)
{
    int e = blockIdx.x * blockDim.x + threadIdx.x;
    if (e >= E_) return;
    int d = dst[e];
    atomicAdd(&deg[d], 1.f);
#pragma unroll
    for (int k = 0; k < 4; ++k) atomicAdd(&lsum[d * 4 + k], eattr[(long)e * 4 + k]);
}
__global__ void k_loop_fin(float* __restrict__ lsum, const float* __restrict__ deg, int n)
{
    int i = blockIdx.x * blockDim.x + threadIdx.x;
    if (i >= n * 4) return;
    lsum[i] /= fmaxf(deg[i >> 2], 1.f);
}

// fold We[4,96] x a_edge[H,C] -> M[4,H]  (collapses edge projection+einsum)
__global__ void k_fold_edge(const float* __restrict__ We, const float* __restrict__ ae,
                            float* __restrict__ Mf)
{
    int i = threadIdx.x;
    if (i < 16) {
        int k = i >> 2, hh = i & 3;
        float s = 0.f;
        for (int c = 0; c < CH; ++c) s += We[k * HID + hh * CH + c] * ae[hh * CH + c];
        Mf[k * 4 + hh] = s;
    }
}

// per-node attention scores a_s, a_d  [N,H]
__global__ void k_att_scores(const float* __restrict__ hp, const float* __restrict__ asrc,
                             const float* __restrict__ adst, float* __restrict__ a_s,
                             float* __restrict__ a_d, int n)
{
    int i = blockIdx.x * blockDim.x + threadIdx.x;
    if (i >= n * HEADS) return;
    int node = i >> 2, hh = i & 3;
    float ss = 0.f, dd = 0.f;
    const float* row = hp + (long)node * HID + hh * CH;
#pragma unroll
    for (int c = 0; c < CH; ++c) {
        float v = row[c];
        ss += v * asrc[hh * CH + c];
        dd += v * adst[hh * CH + c];
    }
    a_s[i] = ss; a_d[i] = dd;
}

// alpha = leakyrelu(a_s[s]+a_d[d]+ea@Mf); segment max via ordered-uint atomicMax
__global__ void k_alpha_max(const int* __restrict__ src, const int* __restrict__ dst,
                            const float* __restrict__ eattr, const float* __restrict__ lattr,
                            const float* __restrict__ a_s, const float* __restrict__ a_d,
                            const float* __restrict__ Mf, float* __restrict__ alpha,
                            unsigned* __restrict__ mx, int E_, int N_)
{
    int i = blockIdx.x * blockDim.x + threadIdx.x;
    if (i >= (E_ + N_) * HEADS) return;
    int idx = i >> 2, hh = i & 3;
    int s, d; const float* ea;
    if (idx < E_) { s = src[idx]; d = dst[idx]; ea = eattr + (long)idx * 4; }
    else          { s = d = idx - E_;           ea = lattr + (long)(idx - E_) * 4; }
    float al = a_s[s * HEADS + hh] + a_d[d * HEADS + hh];
#pragma unroll
    for (int k = 0; k < 4; ++k) al += ea[k] * Mf[k * 4 + hh];
    al = (al > 0.f) ? al : 0.2f * al;
    alpha[i] = al;
    atomicMax(&mx[d * HEADS + hh], fenc(al));
}

// ex = exp(alpha - max[d]); den[d] += ex   (accumulators L2-resident)
__global__ void k_exp_den(const int* __restrict__ dst, float* __restrict__ alpha,
                          const unsigned* __restrict__ mx, float* __restrict__ den,
                          int E_, int N_)
{
    int i = blockIdx.x * blockDim.x + threadIdx.x;
    if (i >= (E_ + N_) * HEADS) return;
    int idx = i >> 2, hh = i & 3;
    int d = (idx < E_) ? dst[idx] : idx - E_;
    float e = __expf(alpha[i] - fdec(mx[d * HEADS + hh]));
    alpha[i] = e;
    atomicAdd(&den[d * HEADS + hh], e);
}

// weighted message scatter: agg[d] += attn * hproj[s]  (24 floats per (edge,head))
__global__ void k_scatter(const int* __restrict__ src, const int* __restrict__ dst,
                          const float* __restrict__ alpha, const float* __restrict__ den,
                          const float* __restrict__ hp, float* __restrict__ agg,
                          int E_, int N_)
{
    int i = blockIdx.x * blockDim.x + threadIdx.x;
    if (i >= (E_ + N_) * HEADS) return;
    int idx = i >> 2, hh = i & 3;
    int s, d;
    if (idx < E_) { s = src[idx]; d = dst[idx]; }
    else          { s = d = idx - E_; }
    float attn = alpha[i] / (den[d * HEADS + hh] + 1e-16f);
    const float* hrow = hp + (long)s * HID + hh * CH;
    float* orow = agg + (long)d * HID + hh * CH;
    __builtin_prefetch((const void*)hrow, 0, 3);   // global_prefetch_b8
#pragma unroll
    for (int c = 0; c < CH; ++c) atomicAdd(&orow[c], attn * hrow[c]);
}

// bias + ELU + residual + LayerNorm; one wave32 per node (3 features/lane)
__global__ void k_epilogue_ln(const float* __restrict__ agg, const float* __restrict__ bias,
                              const float* __restrict__ res, const float* __restrict__ gamma,
                              const float* __restrict__ beta, float* __restrict__ out, int n)
{
    int node = blockIdx.x * (blockDim.x >> 5) + (threadIdx.x >> 5);
    int lane = threadIdx.x & 31;
    if (node >= n) return;
    const long base = (long)node * HID;
    float v[3]; float s = 0.f;
#pragma unroll
    for (int j = 0; j < 3; ++j) {
        int f = lane * 3 + j;
        float t = agg[base + f] + bias[f];
        t = (t > 0.f) ? t : (__expf(t) - 1.f);   // ELU(alpha=1)
        t += res[base + f];
        v[j] = t; s += t;
    }
#pragma unroll
    for (int o = 16; o > 0; o >>= 1) s += __shfl_xor(s, o, 32);
    float mean = s * (1.f / 96.f);
    float vs = 0.f;
#pragma unroll
    for (int j = 0; j < 3; ++j) { float d = v[j] - mean; vs += d * d; }
#pragma unroll
    for (int o = 16; o > 0; o >>= 1) vs += __shfl_xor(vs, o, 32);
    float rstd = rsqrtf(vs * (1.f / 96.f) + 1e-5f);
#pragma unroll
    for (int j = 0; j < 3; ++j) {
        int f = lane * 3 + j;
        out[base + f] = (v[j] - mean) * rstd * gamma[f] + beta[f];
    }
}

__global__ void k_gather(const float* __restrict__ h, const int* __restrict__ zidx,
                         float* __restrict__ z)
{
    int i = blockIdx.x * blockDim.x + threadIdx.x;
    if (i >= NZ * HID) return;
    int r = i / HID, f = i - r * HID;
    z[i] = h[(long)zidx[r] * HID + f];
}

// final tiny layers (K=48, outputs 2/1/1) + concat to [384,4]
__global__ void k_final(const float* __restrict__ tc, const float* __restrict__ td,
                        const float* __restrict__ tj,
                        const float* __restrict__ cw3, const float* __restrict__ cb3,
                        const float* __restrict__ dw3, const float* __restrict__ db3,
                        const float* __restrict__ jw3, const float* __restrict__ jb3,
                        float* __restrict__ out)
{
    int i = blockIdx.x * blockDim.x + threadIdx.x;
    if (i >= NZ * 4) return;
    int r = i >> 2, col = i & 3;
    float s;
    if (col < 2) {
        s = cb3[col];
        for (int k = 0; k < 48; ++k) s += tc[r * 48 + k] * cw3[k * 2 + col];
    } else if (col == 2) {
        s = db3[0];
        for (int k = 0; k < 48; ++k) s += td[r * 48 + k] * dw3[k];
    } else {
        s = jb3[0];
        for (int k = 0; k < 48; ++k) s += tj[r * 48 + k] * jw3[k];
    }
    out[i] = s;
}

extern "C" void kernel_launch(void* const* d_in, const int* in_sizes, int n_in,
                              void* d_out, int out_size, void* d_ws, size_t ws_size,
                              hipStream_t stream)
{
    (void)in_sizes; (void)n_in; (void)out_size; (void)ws_size;
    const float* x        = (const float*)d_in[0];
    const int*   eidx     = (const int*)d_in[1];
    const float* eattr    = (const float*)d_in[2];
    const int*   zidx     = (const int*)d_in[3];
    const float* in_w1    = (const float*)d_in[4];
    const float* in_b1    = (const float*)d_in[5];
    const float* in_w2    = (const float*)d_in[6];
    const float* in_b2    = (const float*)d_in[7];
    const float* conv_w   = (const float*)d_in[8];
    const float* att_src  = (const float*)d_in[9];
    const float* att_dst  = (const float*)d_in[10];
    const float* lin_edge = (const float*)d_in[11];
    const float* att_edge = (const float*)d_in[12];
    const float* conv_b   = (const float*)d_in[13];
    const float* nsc      = (const float*)d_in[14];
    const float* nbi      = (const float*)d_in[15];
    const float* hw[3][6];
    for (int hh = 0; hh < 3; ++hh)
        for (int k = 0; k < 6; ++k) hw[hh][k] = (const float*)d_in[16 + hh * 6 + k];

    const int* src = eidx;
    const int* dst = eidx + N_EDGES;

    float* ws = (float*)d_ws;
    size_t off = 0;
    auto alloc = [&](size_t n) { float* p = ws + off; off += n; return p; };
    float*    hA    = alloc((size_t)N_NODES * HID);
    float*    hB    = alloc((size_t)N_NODES * HID);
    float*    hproj = alloc((size_t)N_NODES * HID);
    float*    agg   = alloc((size_t)N_NODES * HID);
    float*    a_s   = alloc((size_t)N_NODES * HEADS);
    float*    a_d   = alloc((size_t)N_NODES * HEADS);
    float*    alpha = alloc((size_t)(N_EDGES + N_NODES) * HEADS);
    unsigned* mx    = (unsigned*)alloc((size_t)N_NODES * HEADS);
    float*    den   = alloc((size_t)N_NODES * HEADS);
    float*    deg   = alloc((size_t)N_NODES);
    float*    lattr = alloc((size_t)N_NODES * 4);
    float*    Mf    = alloc(16);
    float*    z     = alloc((size_t)NZ * HID);
    float*    t1    = alloc((size_t)NZ * HID);
    float*    t2[3] = { alloc((size_t)NZ * 48), alloc((size_t)NZ * 48), alloc((size_t)NZ * 48) };

    const int TB = 256;
    const int EA = (N_EDGES + N_NODES) * HEADS;
    constexpr int MT = 4;                 // row-tiles per GEMM block

    // ---- layer-invariant: degree + self-loop attr ----
    k_zero_f32<<<cdiv(N_NODES, TB), TB, 0, stream>>>(deg, N_NODES);
    k_zero_f32<<<cdiv(N_NODES * 4, TB), TB, 0, stream>>>(lattr, N_NODES * 4);
    k_deg<<<cdiv(N_EDGES, TB), TB, 0, stream>>>(dst, eattr, deg, lattr, N_EDGES);
    k_loop_fin<<<cdiv(N_NODES * 4, TB), TB, 0, stream>>>(lattr, deg, N_NODES);

    // ---- input MLP ----
    k_in_mlp1<<<cdiv(N_NODES * HID, TB), TB, 0, stream>>>(x, in_w1, in_b1, agg, N_NODES);
    k_gemm_wmma<HID, HID, 0, MT><<<cdiv(N_NODES, 16 * MT), (HID / 16) * 32, 0, stream>>>(
        agg, in_w2, in_b2, hA, N_NODES);

    // ---- 4 GAT layers ----
    float* hcur = hA; float* hnext = hB;
    for (int L = 0; L < 4; ++L) {
        k_gemm_wmma<HID, HID, 0, MT><<<cdiv(N_NODES, 16 * MT), (HID / 16) * 32, 0, stream>>>(
            hcur, conv_w + (size_t)L * HID * HID, nullptr, hproj, N_NODES);
        k_fold_edge<<<1, 32, 0, stream>>>(
            lin_edge + (size_t)L * 4 * HID, att_edge + (size_t)L * HEADS * CH, Mf);
        k_att_scores<<<cdiv(N_NODES * HEADS, TB), TB, 0, stream>>>(
            hproj, att_src + (size_t)L * HEADS * CH, att_dst + (size_t)L * HEADS * CH,
            a_s, a_d, N_NODES);
        k_zero_u32<<<cdiv(N_NODES * HEADS, TB), TB, 0, stream>>>(mx, N_NODES * HEADS);
        k_zero_f32<<<cdiv(N_NODES * HEADS, TB), TB, 0, stream>>>(den, N_NODES * HEADS);
        k_zero_f32<<<cdiv(N_NODES * HID, TB), TB, 0, stream>>>(agg, N_NODES * HID);
        k_alpha_max<<<cdiv(EA, TB), TB, 0, stream>>>(
            src, dst, eattr, lattr, a_s, a_d, Mf, alpha, mx, N_EDGES, N_NODES);
        k_exp_den<<<cdiv(EA, TB), TB, 0, stream>>>(dst, alpha, mx, den, N_EDGES, N_NODES);
        k_scatter<<<cdiv(EA, TB), TB, 0, stream>>>(
            src, dst, alpha, den, hproj, agg, N_EDGES, N_NODES);
        k_epilogue_ln<<<cdiv(N_NODES, TB / 32), TB, 0, stream>>>(
            agg, conv_b + (size_t)L * HID, hcur, nsc + (size_t)L * HID,
            nbi + (size_t)L * HID, hnext, N_NODES);
        float* t = hcur; hcur = hnext; hnext = t;
    }

    // ---- heads ----
    k_gather<<<cdiv(NZ * HID, TB), TB, 0, stream>>>(hcur, zidx, z);
    for (int hh = 0; hh < 3; ++hh) {
        k_gemm_wmma<HID, HID, 1, MT><<<cdiv(NZ, 16 * MT), (HID / 16) * 32, 0, stream>>>(
            z, hw[hh][0], hw[hh][1], t1, NZ);
        k_gemm_wmma<HID, 48, 1, MT><<<cdiv(NZ, 16 * MT), (48 / 16) * 32, 0, stream>>>(
            t1, hw[hh][2], hw[hh][3], t2[hh], NZ);
    }
    k_final<<<cdiv(NZ * 4, TB), TB, 0, stream>>>(
        t2[0], t2[1], t2[2],
        hw[0][4], hw[0][5], hw[1][4], hw[1][5], hw[2][4], hw[2][5],
        (float*)d_out);
}
